// Mamba_mLSTM_8143257993726
// MI455X (gfx1250) — matrix operand order repaired and checked by reading, not compiled, not run on hardware
//
#include <hip/hip_runtime.h>

// Mamba/mLSTM forward for MI455X (gfx1250): wave32, WMMA bf16 f32-acc,
// async global->LDS DMA staging for all matrix tiles.
// Workspace requirement: ~54 MB.

#define SEQ   1024
#define BATCH 4

typedef __attribute__((ext_vector_type(16))) __bf16        v16bf;
typedef __attribute__((ext_vector_type(8)))  float         v8f;
typedef __attribute__((ext_vector_type(8)))  unsigned int  v8u;

__device__ __forceinline__ unsigned short f2bf(float f) {
  unsigned int u = __float_as_uint(f);
  u += 0x7fffu + ((u >> 16) & 1u);          // round-to-nearest-even
  return (unsigned short)(u >> 16);
}

// Async DMA: global -> LDS, 16 bytes per lane (ASYNCcnt-tracked).
__device__ __forceinline__ void async_ld_b128(unsigned lds_addr, const void* sbase,
                                              unsigned voff) {
  asm volatile("global_load_async_to_lds_b128 %0, %1, %2"
               :: "v"(lds_addr), "v"(voff), "s"(sbase) : "memory");
}
__device__ __forceinline__ void wait_async0() {
  asm volatile("s_wait_asynccnt 0x0" ::: "memory");
}
__device__ __forceinline__ unsigned lds_off(const void* p) {
  return (unsigned)(size_t)p;               // low 32 bits of LDS aperture addr
}

// ---------------------------------------------------------------------------
// f32 -> bf16 cast (one-time operand preparation so GEMMs can DMA raw tiles).
// ---------------------------------------------------------------------------
__global__ void k_cast_bf16(const float* __restrict__ in,
                            unsigned short* __restrict__ outp, int n) {
  const int i = blockIdx.x * blockDim.x + threadIdx.x;
  if (i < n) outp[i] = f2bf(in[i]);
}

// ---------------------------------------------------------------------------
// C[M x N] = A[M x K] * B[N x K]^T, bf16 operands, f32 out.
// Block = 128 threads (4 waves), tile 64x64, K-step 32.
// Tiles staged with global_load_async_to_lds_b128.
// ---------------------------------------------------------------------------
__global__ __launch_bounds__(128)
void k_gemm_bf16(const unsigned short* __restrict__ A,
                 const unsigned short* __restrict__ Bw,
                 float* __restrict__ C, int M, int N, int Kd) {
  __shared__ __attribute__((aligned(16))) unsigned short lA[64][32];
  __shared__ __attribute__((aligned(16))) unsigned short lB[64][32];
  const int tid  = threadIdx.x;
  const int lane = tid & 31;
  const int wv   = tid >> 5;
  const int m0   = blockIdx.x * 64;
  const int n0   = blockIdx.y * 64;
  const int lh   = lane & 15;
  const int kbA  = (lane < 16) ? 0 : 8;
  const int kbB  = (lane < 16) ? 0 : 16;
  const int rowA = 16 * wv + lh;
  const unsigned laBase = lds_off(&lA[0][0]);
  const unsigned lbBase = lds_off(&lB[0][0]);

  v8f acc[4];
  for (int t = 0; t < 4; ++t)
    for (int i = 0; i < 8; ++i) acc[t][i] = 0.f;

  for (int k0 = 0; k0 < Kd; k0 += 32) {
    __syncthreads();
    // 64 rows x 64 bytes per tile = 256 x 16B chunks; 2 per thread per matrix.
    for (int rep = 0; rep < 2; ++rep) {
      const int idx = tid + 128 * rep;            // 0..255
      const int r = idx >> 2, c = idx & 3;        // row, 16B chunk in row
      async_ld_b128(laBase + idx * 16, A,
                    (unsigned)(((m0 + r) * Kd + k0) * 2 + c * 16));
      async_ld_b128(lbBase + idx * 16, Bw,
                    (unsigned)(((n0 + r) * Kd + k0) * 2 + c * 16));
    }
    if (k0 + 32 < Kd) {                            // prefetch next K-panel
      __builtin_prefetch(A  + (size_t)(m0 + (tid & 63)) * Kd + k0 + 32, 0, 0);
      __builtin_prefetch(Bw + (size_t)(n0 + (tid & 63)) * Kd + k0 + 32, 0, 0);
    }
    wait_async0();
    __syncthreads();

    v8u a8;
    for (int g = 0; g < 4; ++g) {
      a8[g]     = *(const unsigned int*)&lA[rowA][kbB + 2 * g];
      a8[4 + g] = *(const unsigned int*)&lA[rowA][8 + kbB + 2 * g];
    }
    const v16bf av = __builtin_bit_cast(v16bf, a8);
    for (int t = 0; t < 4; ++t) {
      v8u b8;
      const int colB = 16 * t + lh;
      for (int vi = 0; vi < 8; ++vi)
        b8[vi] = *(const unsigned int*)&lB[colB][kbB + 2 * vi];
      acc[t] = __builtin_amdgcn_wmma_f32_16x16x32_bf16(
          false, av, false, __builtin_bit_cast(v16bf, b8),
          (short)0, acc[t], false, false);
    }
  }
  const int rOff = (lane < 16) ? 0 : 8;
  for (int t = 0; t < 4; ++t)
    for (int r = 0; r < 8; ++r) {
      const int row = m0 + 16 * wv + r + rOff;
      const int col = n0 + 16 * t + lh;
      C[(size_t)row * N + col] = acc[t][r];
    }
}

// ---------------------------------------------------------------------------
// Per (b,s): depthwise causal conv (K=4) + SiLU, block-diagonal 4x4 q/k/v
// projections, and the two 8-head gate GEMVs over concat(q,k,v).
// q,k stored (b,h,s,d) bf16; v stored transposed (b,h,d,s) bf16 so attention
// V-tiles are contiguous for async DMA.
// ---------------------------------------------------------------------------
__global__ __launch_bounds__(256)
void k_conv_qkv_gates(const float* __restrict__ up,
                      const float* __restrict__ conv_w, const float* __restrict__ conv_b,
                      const float* __restrict__ Wq, const float* __restrict__ Wk,
                      const float* __restrict__ Wv,
                      const float* __restrict__ Wi, const float* __restrict__ bi,
                      const float* __restrict__ Wf, const float* __restrict__ bfv,
                      float* __restrict__ xact,
                      unsigned short* __restrict__ qb, unsigned short* __restrict__ kb,
                      unsigned short* __restrict__ vb,
                      float* __restrict__ ig, float* __restrict__ fg) {
  __shared__ float xm[512], xa[512], qL[512], kL[512], vL[512];
  __shared__ float red[256 * 16];
  const int bs  = blockIdx.x;
  const int b   = bs >> 10;
  const int s   = bs & 1023;
  const int tid = threadIdx.x;

  for (int c = tid; c < 512; c += 256) {
    float acc = conv_b[c];
    for (int kk = 0; kk < 4; ++kk) {
      const int sp = s - 3 + kk;
      if (sp >= 0) acc += up[(size_t)(bs - 3 + kk) * 1024 + c] * conv_w[c * 4 + kk];
    }
    const float act = acc / (1.f + __expf(-acc));   // SiLU
    xa[c] = act;
    xact[(size_t)bs * 512 + c] = act;
    xm[c] = up[(size_t)bs * 1024 + c];
  }
  __syncthreads();

  if (tid < 128) {
    const int nb = tid;
    for (int o = 0; o < 4; ++o) {
      float qv = 0.f, kv = 0.f, vv = 0.f;
      for (int d = 0; d < 4; ++d) {
        qv += Wq[nb * 16 + o * 4 + d] * xa[nb * 4 + d];
        kv += Wk[nb * 16 + o * 4 + d] * xa[nb * 4 + d];
        vv += Wv[nb * 16 + o * 4 + d] * xm[nb * 4 + d];
      }
      const int ch = nb * 4 + o;
      qL[ch] = qv; kL[ch] = kv; vL[ch] = vv;
      const int head = ch >> 6, d2 = ch & 63;
      const size_t bh = (size_t)(b * 8 + head);
      qb[(bh * SEQ + s) * 64 + d2] = f2bf(qv);
      kb[(bh * SEQ + s) * 64 + d2] = f2bf(kv * 0.125f);   // fold 1/sqrt(DH)
      vb[(bh * 64 + d2) * SEQ + s] = f2bf(vv);            // transposed
    }
  }
  __syncthreads();

  float pacc[16];
  for (int o = 0; o < 16; ++o) pacc[o] = 0.f;
  for (int j = tid; j < 1536; j += 256) {
    const float g = (j < 512) ? qL[j] : (j < 1024) ? kL[j - 512] : vL[j - 1024];
    for (int hh = 0; hh < 8; ++hh) {
      pacc[hh]     += g * Wi[hh * 1536 + j];
      pacc[8 + hh] += g * Wf[hh * 1536 + j];
    }
  }
  for (int o = 0; o < 16; ++o) red[tid * 16 + o] = pacc[o];
  __syncthreads();
  for (int st = 128; st > 0; st >>= 1) {
    if (tid < st)
      for (int o = 0; o < 16; ++o) red[tid * 16 + o] += red[(tid + st) * 16 + o];
    __syncthreads();
  }
  if (tid < 8) {
    ig[(size_t)(b * 8 + tid) * SEQ + s] = red[tid]     + bi[tid];
    fg[(size_t)(b * 8 + tid) * SEQ + s] = red[8 + tid] + bfv[tid];
  }
}

// ---------------------------------------------------------------------------
// Per (b,h) scan: lf = log_sigmoid(fg), lc = cumsum(lf), a = ig - lc,
// M = cummax(a), em = exp(-(lc + M)). 32 independent scans, one lane each.
// ---------------------------------------------------------------------------
__global__ void k_scan(const float* __restrict__ ig, const float* __restrict__ fg,
                       float* __restrict__ aa, float* __restrict__ mmx,
                       float* __restrict__ em) {
  const int t = threadIdx.x;
  if (t >= 32) return;
  const size_t base = (size_t)t * SEQ;
  float lc = 0.f, amax = -3.4e38f;
  for (int s = 0; s < SEQ; ++s) {
    const float x  = fg[base + s];
    const float lf = fminf(x, 0.f) - log1pf(__expf(-fabsf(x)));
    lc += lf;
    const float av = ig[base + s] - lc;
    amax = fmaxf(amax, av);
    aa [base + s] = av;
    mmx[base + s] = amax;
    em [base + s] = __expf(-(lc + amax));
  }
}

// ---------------------------------------------------------------------------
// Causal decayed attention + per-row normalizer + head LayerNorm.
// Block = 128 threads (4 waves); wave owns 16 query rows; 64-row query tiles.
// K/V tiles DMA'd into LDS with async loads; QK^T and P@V via WMMA bf16;
// row-sum via a ones-column B-fragment.
// ---------------------------------------------------------------------------
__global__ __launch_bounds__(128)
void k_attn(const unsigned short* __restrict__ qbm, const unsigned short* __restrict__ kbm,
            const unsigned short* __restrict__ vbm, const float* __restrict__ aa,
            const float* __restrict__ mmx, const float* __restrict__ em,
            const float* __restrict__ norm_w, float* __restrict__ hbuf) {
  __shared__ __attribute__((aligned(16))) unsigned short Kt[32][64];  // [key][dh]
  __shared__ __attribute__((aligned(16))) unsigned short Vt[64][32];  // [dh][key]
  __shared__ float aT[32];
  __shared__ __attribute__((aligned(16))) unsigned short Pl[4][16][32];

  const int tid  = threadIdx.x;
  const int lane = tid & 31;
  const int wv   = tid >> 5;
  const int qt   = blockIdx.x & 15;
  const int bh   = blockIdx.x >> 4;          // b*8 + head
  const int head = bh & 7;
  const int b    = bh >> 3;

  const unsigned short* qg = qbm + (size_t)bh * SEQ * 64;
  const unsigned short* kg = kbm + (size_t)bh * SEQ * 64;
  const unsigned short* vg = vbm + (size_t)bh * 64 * SEQ;   // [dh][s]
  const float* ag = aa  + (size_t)bh * SEQ;
  const float* mg = mmx + (size_t)bh * SEQ;
  const float* eg = em  + (size_t)bh * SEQ;

  const int q0   = qt * 64 + 16 * wv;
  const int lh   = lane & 15;
  const int rOff = (lane < 16) ? 0 : 8;
  const int kbA  = (lane < 16) ? 0 : 8;
  const int kbB  = (lane < 16) ? 0 : 16;
  const unsigned ktBase = lds_off(&Kt[0][0]);
  const unsigned vtBase = lds_off(&Vt[0][0]);

  int   qidx[8];
  float Mrow[8];
  for (int r = 0; r < 8; ++r) {
    qidx[r] = q0 + r + rOff;
    Mrow[r] = mg[qidx[r]];
  }

  // Q A-fragments (16 rows x 64 dh => two 16x32 chunks), loaded once.
  v16bf qfrag[2];
  {
    const int row = q0 + lh;
    for (int c = 0; c < 2; ++c) {
      v8u a8;
      for (int g = 0; g < 4; ++g) {
        a8[g]     = *(const unsigned int*)(qg + (size_t)row * 64 + 32 * c + kbB + 2 * g);
        a8[4 + g] = *(const unsigned int*)(qg + (size_t)row * 64 + 32 * c + 8 + kbB + 2 * g);
      }
      qfrag[c] = __builtin_bit_cast(v16bf, a8);
    }
  }

  v8f accO[4], accR;
  for (int t = 0; t < 4; ++t)
    for (int i = 0; i < 8; ++i) accO[t][i] = 0.f;
  for (int i = 0; i < 8; ++i) accR[i] = 0.f;

  // Ones B-fragment: column of 1.0 at local n==0 -> row-sum accumulator.
  v8u ones8;
  const unsigned int ov = (lh == 0) ? 0x3f803f80u : 0u;
  for (int vi = 0; vi < 8; ++vi) ones8[vi] = ov;
  const v16bf onesf = __builtin_bit_cast(v16bf, ones8);

  const int njt = (qt + 1) * 2;              // 32-key tiles up to causal limit
  for (int jtile = 0; jtile < njt; ++jtile) {
    const int jt = jtile * 32;
    __syncthreads();
    // K tile: 32 keys x 128B contiguous; V tile: 64 dh-rows x 64B contiguous.
    for (int rep = 0; rep < 2; ++rep) {
      const int idx = tid + 128 * rep;                    // 0..255
      async_ld_b128(ktBase + idx * 16, kg, (unsigned)(jt * 128 + idx * 16));
      const int d = idx >> 2, c4 = idx & 3;
      async_ld_b128(vtBase + idx * 16, vg,
                    (unsigned)((d * SEQ + jt) * 2 + c4 * 16));
    }
    if (tid < 32) aT[tid] = ag[jt + tid];
    wait_async0();
    __syncthreads();

    // Scores for two 16-key halves, apply decay + causal mask, stash P (bf16).
    for (int kt2 = 0; kt2 < 2; ++kt2) {
      v8f sacc;
      for (int i = 0; i < 8; ++i) sacc[i] = 0.f;
      const int keyl = 16 * kt2 + lh;
      for (int c = 0; c < 2; ++c) {
        v8u b8;
        for (int vi = 0; vi < 8; ++vi)
          b8[vi] = *(const unsigned int*)&Kt[keyl][32 * c + kbB + 2 * vi];
        sacc = __builtin_amdgcn_wmma_f32_16x16x32_bf16(
            false, qfrag[c], false, __builtin_bit_cast(v16bf, b8),
            (short)0, sacc, false, false);
      }
      const int   jkey = jt + 16 * kt2 + lh;
      const float aj   = aT[16 * kt2 + lh];
      for (int r = 0; r < 8; ++r) {
        const float p = (jkey <= qidx[r]) ? sacc[r] * __expf(aj - Mrow[r]) : 0.f;
        Pl[wv][r + rOff][16 * kt2 + lh] = f2bf(p);
      }
    }

    // P (16x32) @ V (32x64) accumulation + ones-column row-sum.
    v8u a8;
    for (int g = 0; g < 4; ++g) {
      a8[g]     = *(const unsigned int*)&Pl[wv][lh][kbB + 2 * g];
      a8[4 + g] = *(const unsigned int*)&Pl[wv][lh][8 + kbB + 2 * g];
    }
    const v16bf pf16 = __builtin_bit_cast(v16bf, a8);
    for (int t = 0; t < 4; ++t) {
      v8u b8;
      const int dhn = 16 * t + lh;
      for (int vi = 0; vi < 8; ++vi)
        b8[vi] = *(const unsigned int*)&Vt[dhn][kbB + 2 * vi];
      accO[t] = __builtin_amdgcn_wmma_f32_16x16x32_bf16(
          false, pf16, false, __builtin_bit_cast(v16bf, b8),
          (short)0, accO[t], false, false);
    }
    accR = __builtin_amdgcn_wmma_f32_16x16x32_bf16(
        false, pf16, false, onesf, (short)0, accR, false, false);
  }

  // Epilogue: normalizer, LayerNorm over DH, store f32 (B,S,INNER).
  const int srcl = (lane < 16) ? 0 : 16;
  float nw[4];
  for (int t = 0; t < 4; ++t) nw[t] = norm_w[head * 64 + 16 * t + lh];
  for (int r = 0; r < 8; ++r) {
    const float rs  = __shfl(accR[r], srcl, 32);
    const float nrm = fmaxf(fabsf(rs), eg[qidx[r]]);
    const float inv = 1.f / (nrm + 1e-6f);
    float h0 = accO[0][r] * inv, h1 = accO[1][r] * inv;
    float h2 = accO[2][r] * inv, h3 = accO[3][r] * inv;
    float sm = h0 + h1 + h2 + h3;
    float sq = h0 * h0 + h1 * h1 + h2 * h2 + h3 * h3;
    for (int m = 8; m >= 1; m >>= 1) {
      sm += __shfl_xor(sm, m, 32);
      sq += __shfl_xor(sq, m, 32);
    }
    const float mu  = sm * (1.f / 64.f);
    const float var = sq * (1.f / 64.f) - mu * mu;
    const float sc  = rsqrtf(var + 1e-5f);
    const size_t base = ((size_t)(b * SEQ + qidx[r])) * 512 + head * 64 + lh;
    hbuf[base +  0] = (h0 - mu) * sc * nw[0];
    hbuf[base + 16] = (h1 - mu) * sc * nw[1];
    hbuf[base + 32] = (h2 - mu) * sc * nw[2];
    hbuf[base + 48] = (h3 - mu) * sc * nw[3];
  }
}

// ---------------------------------------------------------------------------
// hbf = bf16( (h + skip*x_act) * silu(z) )  -- operand for the down GEMM.
// ---------------------------------------------------------------------------
__global__ void k_epilogue(const float* __restrict__ hbuf, const float* __restrict__ xact,
                           const float* __restrict__ up, const float* __restrict__ skip,
                           unsigned short* __restrict__ hbf) {
  const size_t idx = (size_t)blockIdx.x * blockDim.x + threadIdx.x;
  if (idx >= (size_t)4096 * 512) return;
  const size_t bs = idx >> 9;
  const int    c  = (int)(idx & 511);
  const float z    = up[bs * 1024 + 512 + c];
  const float silz = z / (1.f + __expf(-z));
  hbf[idx] = f2bf((hbuf[idx] + skip[c] * xact[idx]) * silz);
}

// ---------------------------------------------------------------------------
extern "C" void kernel_launch(void* const* d_in, const int* in_sizes, int n_in,
                              void* d_out, int out_size, void* d_ws, size_t ws_size,
                              hipStream_t stream) {
  (void)in_sizes; (void)n_in; (void)out_size; (void)ws_size;
  const float* x      = (const float*)d_in[0];
  const float* W_up   = (const float*)d_in[1];
  const float* conv_w = (const float*)d_in[2];
  const float* conv_b = (const float*)d_in[3];
  const float* Wq     = (const float*)d_in[4];
  const float* Wk     = (const float*)d_in[5];
  const float* Wv     = (const float*)d_in[6];
  const float* W_i    = (const float*)d_in[7];
  const float* b_i    = (const float*)d_in[8];
  const float* W_f    = (const float*)d_in[9];
  const float* b_f    = (const float*)d_in[10];
  const float* norm_w = (const float*)d_in[11];
  const float* skip   = (const float*)d_in[12];
  const float* W_down = (const float*)d_in[13];
  float* out = (float*)d_out;

  char* ws = (char*)d_ws;                       // offsets (bytes), 256-aligned
  float*          up    = (float*)(ws + 0);           // 4096*1024 f32 = 16 MB
  float*          xact  = (float*)(ws + 16777216);    // 4096*512  f32 =  8 MB
  unsigned short* qb    = (unsigned short*)(ws + 25165824);   // 4 MB
  unsigned short* kb    = (unsigned short*)(ws + 29360128);   // 4 MB
  unsigned short* vb    = (unsigned short*)(ws + 33554432);   // 4 MB
  float*          ig    = (float*)(ws + 37748736);    // 32*1024 f32 each
  float*          fg    = (float*)(ws + 37879808);
  float*          aav   = (float*)(ws + 38010880);
  float*          mmx   = (float*)(ws + 38141952);
  float*          em    = (float*)(ws + 38273024);
  float*          hbuf  = (float*)(ws + 38404096);    // 4096*512 f32 = 8 MB
  unsigned short* xbf   = (unsigned short*)(ws + 46792704);   // 2 MB
  unsigned short* wupbf = (unsigned short*)(ws + 48889856);   // 512 KB
  unsigned short* wdnbf = (unsigned short*)(ws + 49414144);   // 256 KB
  unsigned short* hbf   = (unsigned short*)(ws + 49676288);   // 4 MB

  // 0) cast GEMM operands to bf16 (enables raw async tile DMA)
  hipLaunchKernelGGL(k_cast_bf16, dim3(4096), dim3(256), 0, stream, x, xbf, 1048576);
  hipLaunchKernelGGL(k_cast_bf16, dim3(1024), dim3(256), 0, stream, W_up, wupbf, 262144);
  hipLaunchKernelGGL(k_cast_bf16, dim3(512),  dim3(256), 0, stream, W_down, wdnbf, 131072);
  // 1) up = x @ W_up^T   (4096x256)*(1024x256)^T
  hipLaunchKernelGGL(k_gemm_bf16, dim3(64, 16), dim3(128), 0, stream,
                     xbf, wupbf, up, 4096, 1024, 256);
  // 2) conv + SiLU + blockdiag q/k/v + gates
  hipLaunchKernelGGL(k_conv_qkv_gates, dim3(4096), dim3(256), 0, stream,
                     up, conv_w, conv_b, Wq, Wk, Wv, W_i, b_i, W_f, b_f,
                     xact, qb, kb, vb, ig, fg);
  // 3) per-(b,h) gate scan
  hipLaunchKernelGGL(k_scan, dim3(1), dim3(32), 0, stream, ig, fg, aav, mmx, em);
  // 4) decayed causal attention + layernorm
  hipLaunchKernelGGL(k_attn, dim3(BATCH * 8 * 16), dim3(128), 0, stream,
                     qb, kb, vb, aav, mmx, em, norm_w, hbuf);
  // 5) skip + silu(z) gating -> bf16
  hipLaunchKernelGGL(k_epilogue, dim3((4096 * 512) / 256), dim3(256), 0, stream,
                     hbuf, xact, up, skip, hbf);
  // 6) out = h @ W_down^T   (4096x512)*(256x512)^T
  hipLaunchKernelGGL(k_gemm_bf16, dim3(64, 4), dim3(128), 0, stream,
                     hbf, wdnbf, out, 4096, 256, 512);
}
